// CausalSelfAttention_7954279432839
// MI455X (gfx1250) — compile-verified, matrix operand order checked by reading, not run on hardware
//
#include <hip/hip_runtime.h>

typedef __attribute__((ext_vector_type(16))) _Float16 v16h;
typedef __attribute__((ext_vector_type(8)))  float    v8f;

#define D_MODEL 1024
#define N_HEADS 16
#define HEAD_DIM 64
#define SEQ 2048
#define BATCH 4
#define ROWS (BATCH*SEQ)   // 8192
#define QKV_N (3*D_MODEL)  // 3072

union FragU { v16h v; uint4 q[2]; };

// Load a WMMA A/B fragment (16 halves/lane) from an LDS row.
// Lane's elements: K(i) = 8*hlf + i + 8*(i>>3)  ->  two contiguous 16B groups
// at half-offsets 8*hlf and 16+8*hlf within the row.
static __device__ inline v16h ld_frag(const _Float16* base, int hlf) {
    FragU u;
    u.q[0] = *(const uint4*)(base + 8 * hlf);
    u.q[1] = *(const uint4*)(base + 16 + 8 * hlf);
    return u.v;
}

static __device__ inline v8f wmma16(v16h a, v16h b, v8f c) {
    return __builtin_amdgcn_wmma_f32_16x16x32_f16(
        /*neg_a=*/false, a, /*neg_b=*/false, b,
        /*c_mod=*/(short)0, c, /*reuse_a=*/false, /*reuse_b=*/false);
}

// ---- CDNA5 async memory->LDS DMA (ASYNCcnt-tracked), per ISA ch.10/15.18 ----
// Generic LDS pointers carry the LDS offset in addr[31:0] (flat LDS aperture).
static __device__ inline void async_ld_b128(_Float16* lds_dst, const _Float16* gsrc) {
    unsigned loff = (unsigned)(uintptr_t)lds_dst;
    asm volatile("global_load_async_to_lds_b128 %0, %1, off"
                 :: "v"(loff), "v"(gsrc) : "memory");
}
static __device__ inline void wait_async_0()  { asm volatile("s_wait_asynccnt 0x0"  ::: "memory"); }
static __device__ inline void wait_async_6()  { asm volatile("s_wait_asynccnt 0x6"  ::: "memory"); }
static __device__ inline void wait_async_16() { asm volatile("s_wait_asynccnt 0x10" ::: "memory"); }

// ---------------------------------------------------------------- convert
__global__ __launch_bounds__(256) void cvt_f32_f16_k(const float* __restrict__ in,
                                                     _Float16* __restrict__ out, int n) {
    int i = blockIdx.x * 256 + threadIdx.x;
    if (i < n) out[i] = (_Float16)in[i];
}

// ---------------------------------------------------------------- layernorm
__global__ __launch_bounds__(256) void layernorm_k(const float* __restrict__ x,
                                                   const float* __restrict__ gamma,
                                                   const float* __restrict__ beta,
                                                   _Float16* __restrict__ h) {
    __shared__ float rs[8], rq[8];
    int row  = blockIdx.x;
    int tid  = threadIdx.x;
    int lane = tid & 31, wid = tid >> 5;

    float4 xv = ((const float4*)(x + (size_t)row * D_MODEL))[tid];
    float s = xv.x + xv.y + xv.z + xv.w;
    float q = xv.x * xv.x + xv.y * xv.y + xv.z * xv.z + xv.w * xv.w;
    #pragma unroll
    for (int m = 1; m <= 16; m <<= 1) {
        s += __shfl_xor(s, m, 32);
        q += __shfl_xor(q, m, 32);
    }
    if (lane == 0) { rs[wid] = s; rq[wid] = q; }
    __syncthreads();
    float ts = 0.f, tq = 0.f;
    #pragma unroll
    for (int i = 0; i < 8; ++i) { ts += rs[i]; tq += rq[i]; }
    float mu   = ts * (1.0f / D_MODEL);
    float var  = tq * (1.0f / D_MODEL) - mu * mu;
    float rstd = rsqrtf(var + 1e-5f);

    int c0 = tid * 4;
    float xs[4] = {xv.x, xv.y, xv.z, xv.w};
    #pragma unroll
    for (int j = 0; j < 4; ++j) {
        float v = (xs[j] - mu) * rstd * gamma[c0 + j] + beta[c0 + j];
        h[(size_t)row * D_MODEL + c0 + j] = (_Float16)v;
    }
}

// ---------------------------------------------------------------- GEMM
// C[M,N] = A[M,K] * W[N,K]^T, f16 inputs, f32 accumulate via WMMA.
// Block: 256 threads = 8 waves (4 x 2), tile 128(M) x 64(N), K-stage 64.
// Double-buffered LDS fed by global_load_async_to_lds_b128 (ASYNCcnt);
// steady-state loop is branch-free (last stage peeled).
// mode 0: scatter QKV into q/k/v f16 buffers laid out [B*H][T][64]
// mode 1: write f32 rows [M,1024] to `of`
#define KCH  64  // K halves staged per stage
#define GLDA 72  // LDS row stride in halves (64 data + 8 pad) -> conflict-free
__global__ __launch_bounds__(256) void gemm_f16_k(const _Float16* __restrict__ A,
                                                  const _Float16* __restrict__ W,
                                                  int K, int mode,
                                                  _Float16* __restrict__ oq,
                                                  _Float16* __restrict__ ok,
                                                  _Float16* __restrict__ ov,
                                                  float* __restrict__ of) {
    __shared__ _Float16 sA[2][128 * GLDA];
    __shared__ _Float16 sB[2][64 * GLDA];

    int tid  = threadIdx.x;
    int lane = tid & 31, wid = tid >> 5;
    int lan  = lane & 15, hlf = lane >> 4;
    int wm   = wid & 3,  wn  = wid >> 2;
    int m0   = blockIdx.x * 128;
    int n0   = blockIdx.y * 64;

    v8f acc[2][2] = {};

    // Per-thread async issue of one 64-half K-stage into LDS buffer `buf`
    // (4 + 2 = 6 async-DMA instructions per thread per stage).
    auto issue = [&](int k0, int buf) {
        #pragma unroll
        for (int it = 0; it < 4; ++it) {          // A: 128 rows x 64 halves
            int u = tid + 256 * it;
            int row = u >> 3, seg = u & 7;
            async_ld_b128(&sA[buf][row * GLDA + seg * 8],
                          &A[(size_t)(m0 + row) * K + k0 + seg * 8]);
        }
        #pragma unroll
        for (int it = 0; it < 2; ++it) {          // B: 64 rows x 64 halves
            int u = tid + 256 * it;
            int row = u >> 3, seg = u & 7;
            async_ld_b128(&sB[buf][row * GLDA + seg * 8],
                          &W[(size_t)(n0 + row) * K + k0 + seg * 8]);
        }
    };
    // One stage of compute: 2 K-steps of 32 halves -> 8 WMMAs
    auto compute = [&](int buf) {
        #pragma unroll
        for (int kh = 0; kh < KCH; kh += 32) {
            v16h a0 = ld_frag(&sA[buf][(wm * 32 + 0  + lan) * GLDA + kh], hlf);
            v16h a1 = ld_frag(&sA[buf][(wm * 32 + 16 + lan) * GLDA + kh], hlf);
            v16h b0 = ld_frag(&sB[buf][(wn * 32 + 0  + lan) * GLDA + kh], hlf);
            v16h b1 = ld_frag(&sB[buf][(wn * 32 + 16 + lan) * GLDA + kh], hlf);
            acc[0][0] = wmma16(a0, b0, acc[0][0]);
            acc[0][1] = wmma16(a0, b1, acc[0][1]);
            acc[1][0] = wmma16(a1, b0, acc[1][0]);
            acc[1][1] = wmma16(a1, b1, acc[1][1]);
        }
    };

    int nk = K / KCH;
    issue(0, 0);
    int i = 0;
    for (; i < nk - 1; ++i) {            // branch-free steady state
        issue((i + 1) * KCH, (i ^ 1) & 1);  // prefetch next stage into other buffer
        wait_async_6();                  // stage i's 6 DMAs retired (in-order)
        __syncthreads();                 // all waves' DMA data visible
        compute(i & 1);
        __syncthreads();                 // protect buffer before refill
    }
    wait_async_0();                      // peeled last stage
    __syncthreads();
    compute(i & 1);

    #pragma unroll
    for (int mi = 0; mi < 2; ++mi)
        #pragma unroll
        for (int ni = 0; ni < 2; ++ni)
            #pragma unroll
            for (int r = 0; r < 8; ++r) {
                float v  = acc[mi][ni][r];
                int  m_g = m0 + wm * 32 + mi * 16 + r + 8 * hlf;
                int  n_g = n0 + wn * 32 + ni * 16 + lan;
                if (mode == 0) {
                    int which = n_g >> 10, rem = n_g & 1023;
                    int head  = rem >> 6,  d   = rem & 63;
                    int b = m_g >> 11, t = m_g & 2047;
                    size_t dst = ((size_t)(b * N_HEADS + head) * SEQ + t) * HEAD_DIM + d;
                    _Float16 hv = (_Float16)v;
                    if (which == 0)      oq[dst] = hv;
                    else if (which == 1) ok[dst] = hv;
                    else                 ov[dst] = hv;
                } else {
                    of[(size_t)m_g * D_MODEL + n_g] = v;
                }
            }
}

// ---------------------------------------------------------------- attention
// One wave per (16-query tile, head). Flash-attention over 32-key chunks,
// K/V double-buffered via async DMA-to-LDS; last chunk peeled.
#define ALD 72   // K/V/Q LDS row stride in halves (64 data + 8 pad)
#define PLD 40   // P LDS row stride in halves (32 data + 8 pad)
__global__ __launch_bounds__(32) void attn_k(const _Float16* __restrict__ Qm,
                                             const _Float16* __restrict__ Km,
                                             const _Float16* __restrict__ Vm,
                                             _Float16* __restrict__ O) {
    __shared__ _Float16 sQ[16 * ALD];
    __shared__ _Float16 sK[2][32 * ALD];
    __shared__ _Float16 sV[2][32 * ALD];
    __shared__ _Float16 sP[16 * PLD];

    int lane = threadIdx.x;
    int lan  = lane & 15, hlf = lane >> 4;
    int qt   = blockIdx.x, bh = blockIdx.y;
    int q0   = qt * 16;

    const _Float16* Qb = Qm + ((size_t)bh * SEQ + q0) * HEAD_DIM;
    const _Float16* Kb = Km + (size_t)bh * SEQ * HEAD_DIM;
    const _Float16* Vb = Vm + (size_t)bh * SEQ * HEAD_DIM;

    // K/V chunk issue: 32x64 halves each = 8+8 async DMAs per lane
    auto issueKV = [&](int c0, int buf) {
        #pragma unroll
        for (int it = 0; it < 8; ++it) {
            int u = lane + 32 * it;
            int row = u >> 3, seg = u & 7;
            async_ld_b128(&sK[buf][row * ALD + seg * 8],
                          &Kb[(size_t)(c0 + row) * HEAD_DIM + seg * 8]);
            async_ld_b128(&sV[buf][row * ALD + seg * 8],
                          &Vb[(size_t)(c0 + row) * HEAD_DIM + seg * 8]);
        }
    };

    // Chunk 0 + Q tile in flight together, then wait once.
    issueKV(0, 0);
    #pragma unroll
    for (int it = 0; it < 4; ++it) {
        int u = lane + 32 * it;
        int row = u >> 3, seg = u & 7;
        async_ld_b128(&sQ[row * ALD + seg * 8], &Qb[row * HEAD_DIM + seg * 8]);
    }
    wait_async_0();

    // Q fragments resident in registers for the whole kernel (K halves 0..31, 32..63)
    v16h qa0 = ld_frag(&sQ[lan * ALD + 0],  hlf);
    v16h qa1 = ld_frag(&sQ[lan * ALD + 32], hlf);

    v8f   acc[4] = {};
    float mr[8], lr[8];
    #pragma unroll
    for (int r = 0; r < 8; ++r) { mr[r] = -3.0e38f; lr[r] = 0.f; }

    auto compute = [&](int kc, int buf) {
        int c0 = kc * 32;
        const _Float16* sKb = sK[buf];
        const _Float16* sVb = sV[buf];

        // S = Q*K^T (scaled, causal-masked): two 16x16 n-tiles
        float s0[8], s1[8];
        #pragma unroll
        for (int kn = 0; kn < 2; ++kn) {
            v16h kb0 = ld_frag(&sKb[(kn * 16 + lan) * ALD + 0],  hlf);
            v16h kb1 = ld_frag(&sKb[(kn * 16 + lan) * ALD + 32], hlf);
            v8f c = {};
            c = wmma16(qa0, kb0, c);
            c = wmma16(qa1, kb1, c);
            float* dst = kn ? s1 : s0;
            #pragma unroll
            for (int r = 0; r < 8; ++r) {
                float sv  = c[r] * 0.125f;              // 1/sqrt(64)
                int   m_g = q0 + r + 8 * hlf;
                int   n_g = c0 + kn * 16 + lan;
                dst[r] = (n_g > m_g) ? -3.0e38f : sv;
            }
        }
        // Online softmax: row stats live inside each 16-lane half (WMMA C layout)
        #pragma unroll
        for (int r = 0; r < 8; ++r) {
            float v = fmaxf(s0[r], s1[r]);
            v = fmaxf(v, __shfl_xor(v, 1, 32));
            v = fmaxf(v, __shfl_xor(v, 2, 32));
            v = fmaxf(v, __shfl_xor(v, 4, 32));
            v = fmaxf(v, __shfl_xor(v, 8, 32));
            float mnew  = fmaxf(mr[r], v);
            float alpha = __expf(mr[r] - mnew);
            mr[r] = mnew;
            lr[r] *= alpha;
            #pragma unroll
            for (int nt = 0; nt < 4; ++nt) acc[nt][r] *= alpha;

            float p0 = __expf(s0[r] - mnew);
            float p1 = __expf(s1[r] - mnew);
            sP[(r + 8 * hlf) * PLD + lan]      = (_Float16)p0;
            sP[(r + 8 * hlf) * PLD + 16 + lan] = (_Float16)p1;
            float ps = p0 + p1;
            ps += __shfl_xor(ps, 1, 32);
            ps += __shfl_xor(ps, 2, 32);
            ps += __shfl_xor(ps, 4, 32);
            ps += __shfl_xor(ps, 8, 32);
            lr[r] += ps;
        }
        // acc += P * V  (P: 16x32 A-fragment from LDS; V gathered per-lane;
        // DS ops are in-order within the wave, so P stores are visible)
        v16h pa = ld_frag(&sP[lan * PLD], hlf);
        #pragma unroll
        for (int nt = 0; nt < 4; ++nt) {
            v16h vb;
            #pragma unroll
            for (int i = 0; i < 16; ++i) {
                int kk = 8 * hlf + i + 8 * (i >> 3);   // K(i) mapping
                vb[i] = sVb[kk * ALD + nt * 16 + lan];
            }
            acc[nt] = wmma16(pa, vb, acc[nt]);
        }
    };

    int nch = (q0 + 15) / 32 + 1;  // key chunks needed under causal mask
    int kc = 0;
    for (; kc < nch - 1; ++kc) {          // branch-free steady state
        issueKV((kc + 1) * 32, (kc ^ 1) & 1);
        wait_async_16();                  // chunk kc's 16 DMAs retired (in-order)
        compute(kc, kc & 1);
    }
    compute_last:
    wait_async_0();                       // peeled final chunk
    compute(kc, kc & 1);

    // Epilogue: O laid out [B*T, D_MODEL] with head offset, f16 for out-proj GEMM
    int b = bh >> 4, hd = bh & 15;
    #pragma unroll
    for (int nt = 0; nt < 4; ++nt)
        #pragma unroll
        for (int r = 0; r < 8; ++r) {
            float o = acc[nt][r] / lr[r];
            int t = q0 + r + 8 * hlf;
            int d = hd * HEAD_DIM + nt * 16 + lan;
            O[((size_t)b * SEQ + t) * D_MODEL + d] = (_Float16)o;
        }
}

// ---------------------------------------------------------------- launch
extern "C" void kernel_launch(void* const* d_in, const int* in_sizes, int n_in,
                              void* d_out, int out_size, void* d_ws, size_t ws_size,
                              hipStream_t stream) {
    const float* x     = (const float*)d_in[0];
    const float* gamma = (const float*)d_in[1];
    const float* beta  = (const float*)d_in[2];
    const float* wqkv  = (const float*)d_in[3];
    const float* wout  = (const float*)d_in[4];
    float* out = (float*)d_out;

    // Workspace partition (bytes), all offsets 256-aligned
    char* ws = (char*)d_ws;
    size_t off = 0;
    _Float16* hf    = (_Float16*)(ws + off); off += (size_t)ROWS * D_MODEL * 2;       // 16 MB
    _Float16* wqkvh = (_Float16*)(ws + off); off += (size_t)QKV_N * D_MODEL * 2;      // 6 MB
    _Float16* wouth = (_Float16*)(ws + off); off += (size_t)D_MODEL * D_MODEL * 2;    // 2 MB
    _Float16* qb    = (_Float16*)(ws + off); off += (size_t)ROWS * D_MODEL * 2;       // 16 MB
    _Float16* kb    = (_Float16*)(ws + off); off += (size_t)ROWS * D_MODEL * 2;       // 16 MB
    _Float16* vb    = (_Float16*)(ws + off); off += (size_t)ROWS * D_MODEL * 2;       // 16 MB
    _Float16* ao    = (_Float16*)(ws + off); off += (size_t)ROWS * D_MODEL * 2;       // 16 MB
    (void)ws_size; (void)n_in; (void)in_sizes; (void)out_size;

    // 1) weights -> f16
    int nwq = QKV_N * D_MODEL, nwo = D_MODEL * D_MODEL;
    cvt_f32_f16_k<<<(nwq + 255) / 256, 256, 0, stream>>>(wqkv, wqkvh, nwq);
    cvt_f32_f16_k<<<(nwo + 255) / 256, 256, 0, stream>>>(wout, wouth, nwo);

    // 2) LayerNorm -> h (f16)
    layernorm_k<<<ROWS, 256, 0, stream>>>(x, gamma, beta, hf);

    // 3) QKV = h @ Wqkv^T, scattered to [B*H][T][64] f16
    gemm_f16_k<<<dim3(ROWS / 128, QKV_N / 64), 256, 0, stream>>>(
        hf, wqkvh, D_MODEL, /*mode=*/0, qb, kb, vb, nullptr);

    // 4) Causal attention, one wave per (16-query tile, head)
    attn_k<<<dim3(SEQ / 16, BATCH * N_HEADS), 32, 0, stream>>>(qb, kb, vb, ao);

    // 5) out = attn_out @ Wout^T (f32 to d_out)
    gemm_f16_k<<<dim3(ROWS / 128, D_MODEL / 64), 256, 0, stream>>>(
        ao, wouth, D_MODEL, /*mode=*/1, nullptr, nullptr, nullptr, out);
}
